// LSTM_Net_12412455486081
// MI455X (gfx1250) — matrix operand order repaired and checked by reading, not compiled, not run on hardware
//
#include <hip/hip_runtime.h>

// ---------------- types ----------------
typedef __bf16 bf16;
typedef __attribute__((ext_vector_type(16))) __bf16 v16bf;
typedef __attribute__((ext_vector_type(8)))  __bf16 v8bf;
typedef __attribute__((ext_vector_type(8)))  float  v8f;

#define NB   128   // batch
#define NT   512   // time steps
#define DIN  128
#define HH   1024
#define DOUT 128

// ---------------- math helpers ----------------
__device__ __forceinline__ float sigf(float x)  { return 1.0f / (1.0f + __expf(-x)); }
__device__ __forceinline__ float tanhf_(float x){ return 1.0f - 2.0f / (__expf(2.0f * x) + 1.0f); }

// ---------------- WMMA fragment loaders (ISA 7.12.2, wave32) ----------------
// A fragment, 16x32 bf16, row-major source with leading dim ld (elements).
// lane L<16: row M=L, K = k..k+7 (elems 0..7) and k+16..k+23 (elems 8..15)
// lane L>=16: row M=L-16, K = k+8..k+15 and k+24..k+31
__device__ __forceinline__ v16bf load_a(const bf16* __restrict__ base, int ld, int k, int lane) {
    int r  = lane & 15;
    int ko = (lane >> 4) << 3;           // 0 or 8
    const bf16* p = base + (size_t)r * ld + k + ko;
    v8bf lo = *(const v8bf*)(p);
    v8bf hi = *(const v8bf*)(p + 16);
    v16bf f;
#pragma unroll
    for (int i = 0; i < 8; ++i) { f[i] = lo[i]; f[i + 8] = hi[i]; }
    return f;
}

// B fragment, 32x16 bf16 (K x N). B[k][n] = W[n][k] with W row-major [N, K], ld = K.
// lane L<16: column N=L, K = k..k+7 and k+16..k+23 ; lane L>=16: column N=L-16, K = k+8..k+15 and k+24..k+31
__device__ __forceinline__ v16bf load_b(const bf16* __restrict__ base, int ld, int k, int lane) {
    int r  = lane & 15;
    int ko = (lane >> 4) << 3;           // 0 or 8
    const bf16* p = base + (size_t)r * ld + k + ko;
    v8bf lo = *(const v8bf*)(p);
    v8bf hi = *(const v8bf*)(p + 16);
    v16bf f;
#pragma unroll
    for (int i = 0; i < 8; ++i) { f[i] = lo[i]; f[i + 8] = hi[i]; }
    return f;
}

// ---------------- fused gates-GEMM + LSTM cell kernel ----------------
// grid: 32 blocks (hidden slices of 32), block: 256 threads = 8 waves.
// Wave w: hidden tile n0 = 32*blk + 16*(w>>2), batch rows [32*(w&3), 32*(w&3)+32)
// as 2 M-tiles. Wave owns 2x4 accumulators = {i,f,g,o} gate tiles for both M-tiles;
// each B (weight) fragment is reused across the 2 M-tiles (1.5 loads per WMMA).
// gates = A1 @ W1^T + A2 @ W2^T + bi + bh ; then LSTM cell update in-register.
__global__ __launch_bounds__(256)
void lstm_layer_kernel(const bf16* __restrict__ A1, int ldA1, int K1,
                       const bf16* __restrict__ W1,           // [4H, K1], ld = K1
                       const bf16* __restrict__ A2,           // [NB, HH], ld = HH (h_prev)
                       const bf16* __restrict__ W2,           // [4H, HH], ld = HH
                       const float* __restrict__ bi,
                       const float* __restrict__ bh,
                       float* __restrict__ c,                 // [NB, HH] f32, in/out
                       bf16*  __restrict__ hout)              // [NB, HH] bf16, out
{
    const int lane = threadIdx.x & 31;
    const int wave = threadIdx.x >> 5;
    const int n0   = blockIdx.x * 32 + ((wave >> 2) << 4);  // hidden-unit tile (16 wide)
    const int m0   = (wave & 3) << 5;                       // batch rows, 2 tiles of 16

    v8f acc[2][4];
#pragma unroll
    for (int i = 0; i < 2; ++i)
#pragma unroll
        for (int g = 0; g < 4; ++g) acc[i][g] = 0.0f;

    // phase 1: input contribution (x_t or h1_new), K = K1
    {
        const bf16* a1 = A1 + (size_t)m0 * ldA1;
        for (int k = 0; k < K1; k += 32) {
            v16bf a0 = load_a(a1, ldA1, k, lane);
            v16bf a1f = load_a(a1 + (size_t)16 * ldA1, ldA1, k, lane);
#pragma unroll
            for (int g = 0; g < 4; ++g) {
                v16bf b = load_b(W1 + (size_t)(n0 + g * HH) * K1, K1, k, lane);
                acc[0][g] = __builtin_amdgcn_wmma_f32_16x16x32_bf16(
                    false, a0,  false, b, (short)0, acc[0][g], false, false);
                acc[1][g] = __builtin_amdgcn_wmma_f32_16x16x32_bf16(
                    false, a1f, false, b, (short)0, acc[1][g], false, false);
            }
        }
    }
    // phase 2: recurrent contribution h_prev @ W2^T, K = HH
    {
        const bf16* a2 = A2 + (size_t)m0 * HH;
        for (int k = 0; k < HH; k += 32) {
            v16bf a0 = load_a(a2, HH, k, lane);
            v16bf a1f = load_a(a2 + (size_t)16 * HH, HH, k, lane);
#pragma unroll
            for (int g = 0; g < 4; ++g) {
                v16bf b = load_b(W2 + (size_t)(n0 + g * HH) * HH, HH, k, lane);
                acc[0][g] = __builtin_amdgcn_wmma_f32_16x16x32_bf16(
                    false, a0,  false, b, (short)0, acc[0][g], false, false);
                acc[1][g] = __builtin_amdgcn_wmma_f32_16x16x32_bf16(
                    false, a1f, false, b, (short)0, acc[1][g], false, false);
            }
        }
    }

    // biases per gate column
    const int n = n0 + (lane & 15);
    float bsum[4];
#pragma unroll
    for (int g = 0; g < 4; ++g) bsum[g] = bi[n + g * HH] + bh[n + g * HH];

    // cell update: accumulators map (vgpr v, lane) -> row m0+16*i+v(+8 for lanes>=16), col n
    const int roff = (lane >> 4) << 3;
#pragma unroll
    for (int i = 0; i < 2; ++i) {
#pragma unroll
        for (int v = 0; v < 8; ++v) {
            const int m = m0 + 16 * i + v + roff;
            const size_t idx = (size_t)m * HH + n;
            const float iv = sigf  (acc[i][0][v] + bsum[0]);
            const float fv = sigf  (acc[i][1][v] + bsum[1]);
            const float gv = tanhf_(acc[i][2][v] + bsum[2]);
            const float ov = sigf  (acc[i][3][v] + bsum[3]);
            const float cn = fv * c[idx] + iv * gv;
            c[idx]    = cn;
            hout[idx] = (bf16)(ov * tanhf_(cn));
        }
    }
}

// ---------------- projection GEMM: out = A @ W^T + bias ----------------
// grid: 8 blocks (N tiles), 8 waves (M tiles). N = 128 fixed.
// If ybf != nullptr: write bf16 to ybf [128,128]. Else write f32 to fout with row stride.
__global__ __launch_bounds__(256)
void proj_kernel(const bf16* __restrict__ A, int ldA, int K,
                 const bf16* __restrict__ W,            // [128, K]
                 const float* __restrict__ bias,
                 bf16*  __restrict__ ybf,
                 float* __restrict__ fout, long frow)
{
    const int lane = threadIdx.x & 31;
    const int wave = threadIdx.x >> 5;
    const int n0   = blockIdx.x * 16;
    const int m0   = wave * 16;

    v8f acc = 0.0f;
    const bf16* a0 = A + (size_t)m0 * ldA;
    const bf16* w0 = W + (size_t)n0 * K;
    for (int k = 0; k < K; k += 32) {
        v16bf a = load_a(a0, ldA, k, lane);
        v16bf b = load_b(w0, K, k, lane);
        acc = __builtin_amdgcn_wmma_f32_16x16x32_bf16(
            false, a, false, b, (short)0, acc, false, false);
    }

    const int n    = n0 + (lane & 15);
    const float bv = bias[n];
    const int roff = (lane >> 4) << 3;
#pragma unroll
    for (int v = 0; v < 8; ++v) {
        const int m = m0 + v + roff;
        const float val = acc[v] + bv;
        if (ybf) ybf[(size_t)m * DOUT + n] = (bf16)val;
        else     fout[(size_t)m * frow + n] = val;
    }
}

// ---------------- f32 -> bf16 conversion ----------------
__global__ void cvt_bf16_kernel(const float* __restrict__ in, bf16* __restrict__ out, long nelem) {
    long i = (long)blockIdx.x * blockDim.x + threadIdx.x;
    const long stride = (long)gridDim.x * blockDim.x;
    for (; i < nelem; i += stride) out[i] = (bf16)in[i];
}

// ---------------- host launcher ----------------
extern "C" void kernel_launch(void* const* d_in, const int* in_sizes, int n_in,
                              void* d_out, int out_size, void* d_ws, size_t ws_size,
                              hipStream_t stream) {
    const float* x     = (const float*)d_in[0];
    const float* W_ih1 = (const float*)d_in[1];
    const float* W_hh1 = (const float*)d_in[2];
    const float* b_ih1 = (const float*)d_in[3];
    const float* b_hh1 = (const float*)d_in[4];
    const float* W_ih2 = (const float*)d_in[5];
    const float* W_hh2 = (const float*)d_in[6];
    const float* b_ih2 = (const float*)d_in[7];
    const float* b_hh2 = (const float*)d_in[8];
    const float* W_lin = (const float*)d_in[9];
    const float* b_lin = (const float*)d_in[10];
    const float* W_act = (const float*)d_in[11];
    const float* b_act = (const float*)d_in[12];
    // d_in[13] = future (0 in this configuration)

    // ---- carve workspace (256B aligned regions) ----
    char* w = (char*)d_ws;
    size_t off = 0;
    auto carve = [&](size_t bytes) -> char* {
        char* p = w + off;
        off += (bytes + 255) & ~(size_t)255;
        return p;
    };
    bf16* xbf   = (bf16*)carve((size_t)NB * NT * DIN * 2);   // 16 MB
    bf16* wih1  = (bf16*)carve((size_t)4 * HH * DIN * 2);    // 1 MB
    bf16* whh1  = (bf16*)carve((size_t)4 * HH * HH * 2);     // 8 MB
    bf16* wih2  = (bf16*)carve((size_t)4 * HH * HH * 2);     // 8 MB
    bf16* whh2  = (bf16*)carve((size_t)4 * HH * HH * 2);     // 8 MB
    bf16* wlin  = (bf16*)carve((size_t)DOUT * HH * 2);
    bf16* wact  = (bf16*)carve((size_t)DOUT * DIN * 2);
    bf16* h1b[2]; h1b[0] = (bf16*)carve((size_t)NB * HH * 2);
                  h1b[1] = (bf16*)carve((size_t)NB * HH * 2);
    bf16* h2b[2]; h2b[0] = (bf16*)carve((size_t)NB * HH * 2);
                  h2b[1] = (bf16*)carve((size_t)NB * HH * 2);
    float* c1   = (float*)carve((size_t)NB * HH * 4);
    float* c2   = (float*)carve((size_t)NB * HH * 4);
    bf16* ybf   = (bf16*)carve((size_t)NB * DOUT * 2);
    (void)ws_size;

    // ---- one-time (per call) conversions to bf16 ----
    cvt_bf16_kernel<<<2048, 256, 0, stream>>>(x,     xbf,  (long)NB * NT * DIN);
    cvt_bf16_kernel<<< 512, 256, 0, stream>>>(W_ih1, wih1, (long)4 * HH * DIN);
    cvt_bf16_kernel<<<2048, 256, 0, stream>>>(W_hh1, whh1, (long)4 * HH * HH);
    cvt_bf16_kernel<<<2048, 256, 0, stream>>>(W_ih2, wih2, (long)4 * HH * HH);
    cvt_bf16_kernel<<<2048, 256, 0, stream>>>(W_hh2, whh2, (long)4 * HH * HH);
    cvt_bf16_kernel<<< 128, 256, 0, stream>>>(W_lin, wlin, (long)DOUT * HH);
    cvt_bf16_kernel<<<  64, 256, 0, stream>>>(W_act, wact, (long)DOUT * DIN);

    // ---- zero initial state (0x0000 is +0 for both bf16 and f32) ----
    hipMemsetAsync(h1b[0], 0, (size_t)NB * HH * 2, stream);
    hipMemsetAsync(h1b[1], 0, (size_t)NB * HH * 2, stream);
    hipMemsetAsync(h2b[0], 0, (size_t)NB * HH * 2, stream);
    hipMemsetAsync(h2b[1], 0, (size_t)NB * HH * 2, stream);
    hipMemsetAsync(c1,     0, (size_t)NB * HH * 4, stream);
    hipMemsetAsync(c2,     0, (size_t)NB * HH * 4, stream);

    float* outp = (float*)d_out;   // [NB, NT, DOUT]

    // ---- sequential recurrence ----
    for (int t = 0; t < NT; ++t) {
        const int p = t & 1;
        const bf16* h1p = h1b[p];       bf16* h1n = h1b[1 - p];
        const bf16* h2p = h2b[p];       bf16* h2n = h2b[1 - p];

        // layer 1: gates = x_t @ W_ih1^T + h1 @ W_hh1^T + b ; cell update -> h1n, c1
        lstm_layer_kernel<<<HH / 32, 256, 0, stream>>>(
            xbf + (size_t)t * DIN, NT * DIN, DIN, wih1,
            h1p, whh1, b_ih1, b_hh1, c1, h1n);

        // layer 2: gates = h1n @ W_ih2^T + h2 @ W_hh2^T + b ; cell update -> h2n, c2
        lstm_layer_kernel<<<HH / 32, 256, 0, stream>>>(
            h1n, HH, HH, wih2,
            h2p, whh2, b_ih2, b_hh2, c2, h2n);

        // y = h2n @ W_lin^T + b_lin  (bf16 intermediate)
        proj_kernel<<<DOUT / 16, 256, 0, stream>>>(
            h2n, HH, HH, wlin, b_lin, ybf, nullptr, 0);

        // out_t = y @ W_act^T + b_act  (f32 final output)
        proj_kernel<<<DOUT / 16, 256, 0, stream>>>(
            ybf, DOUT, DIN, wact, b_act, nullptr,
            outp + (size_t)t * DOUT, (long)NT * DOUT);
    }
    (void)in_sizes; (void)n_in; (void)out_size;
}
